// GVP_GNN_Structural_Stream_41515153883229
// MI455X (gfx1250) — matrix operand order repaired and checked by reading, not compiled, not run on hardware
//
#include <hip/hip_runtime.h>

// ---------------------------------------------------------------------------
// GVP-GNN structural stream for MI455X (gfx1250, wave32, WMMA)
// ---------------------------------------------------------------------------

typedef _Float16 h16;
typedef __attribute__((ext_vector_type(16))) _Float16 v16h;
typedef __attribute__((ext_vector_type(8)))  _Float16 v8h;
typedef __attribute__((ext_vector_type(8)))  float    v8f;

#define NN   20000
#define NE   320000
#define NGR  32
#define ESM  1280
#define GEO  15
#define H    128
#define OUTD 256
#define KNODE 1312   // 1289 padded to 41*32
#define KMSG  384
#define KUPD  256

// ---------------- WMMA helpers (16x16x32 f16 -> f32) -----------------------
// A operand k order used here (contiguous, matched by ldB): lanes 0-15 & 16-31 both hold rows M=0..15.
//   lanes 0-15 : halves[0..15]=K(koff..koff+15)
//   lanes 16-31: halves[0..15]=K(koff+16..+31)
__device__ __forceinline__ v16h ldA(const h16* __restrict__ row, int koff, int hi) {
  return *(const v16h*)(row + koff + hi * 16);
}
// B operand k order used here (contiguous, same as ldA): lane n (0-15) halves[0..15]=K(koff..koff+15);
// lanes 16-31 same N, K(koff+16..+31). Weights stored K-major: W[n][k].
__device__ __forceinline__ v16h ldB(const h16* __restrict__ w, int n, int Kd,
                                    int koff, int hi) {
  return *(const v16h*)(w + (size_t)n * Kd + koff + hi * 16);
}
__device__ __forceinline__ v8f wmma32(v16h a, v16h b, v8f c) {
  return __builtin_amdgcn_wmma_f32_16x16x32_f16(false, a, false, b, (short)0, c,
                                                false, false);
}
__device__ __forceinline__ float gelu_exact(float x) {
  return 0.5f * x * (1.0f + erff(x * 0.70710678118654752f));
}
__device__ __forceinline__ float sigm(float x) { return 1.0f / (1.0f + expf(-x)); }

// ---------------- small utility kernels ------------------------------------
__global__ void k_zero(float* __restrict__ p, int n) {
  int i = blockIdx.x * blockDim.x + threadIdx.x;
  if (i < n) p[i] = 0.0f;
}

__global__ void k_pack(const float* __restrict__ s, h16* __restrict__ d,
                       int O, int Ks, int Kd) {
  int idx = blockIdx.x * blockDim.x + threadIdx.x;
  if (idx >= O * Kd) return;
  int o = idx / Kd, k = idx - o * Kd;
  d[idx] = (h16)(k < Ks ? s[(size_t)o * Ks + k] : 0.0f);
}

// ---------------- input staging --------------------------------------------
__global__ void k_build_node(const float* __restrict__ esm,
                             const float* __restrict__ geo,
                             const float* __restrict__ coords,
                             h16* __restrict__ nin, float* __restrict__ nv) {
  const int i = blockIdx.x;
  const int tid = threadIdx.x;
  for (int k = tid; k < KNODE; k += blockDim.x) {
    float v = 0.0f;
    if (k < ESM)            v = esm[(size_t)i * ESM + k];
    else if (k < ESM + 9)   v = geo[(size_t)i * GEO + 6 + (k - ESM)];
    nin[(size_t)i * KNODE + k] = (h16)v;
  }
  if (tid == 0) {
    float ca[3], cb[3], v0[3], dv[3], nr[3], tv[3];
#pragma unroll
    for (int c = 0; c < 3; ++c) { ca[c] = coords[(size_t)i * 3 + c]; cb[c] = geo[(size_t)i * GEO + 3 + c]; }
#pragma unroll
    for (int c = 0; c < 3; ++c) v0[c] = cb[c] - ca[c];
    {
      float l = sqrtf(v0[0]*v0[0]+v0[1]*v0[1]+v0[2]*v0[2]);
      float inv = 1.0f / fmaxf(l, 1e-12f);
      for (int c = 0; c < 3; ++c) v0[c] *= inv;
    }
    if (i < NN - 1) { for (int c = 0; c < 3; ++c) dv[c] = coords[(size_t)(i+1)*3+c] - ca[c]; }
    else            { for (int c = 0; c < 3; ++c) dv[c] = 0.0f; }
    {
      float l = sqrtf(dv[0]*dv[0]+dv[1]*dv[1]+dv[2]*dv[2]);
      float inv = 1.0f / fmaxf(l, 1e-12f);
      for (int c = 0; c < 3; ++c) dv[c] *= inv;
    }
    nr[0] = v0[1]*dv[2] - v0[2]*dv[1];
    nr[1] = v0[2]*dv[0] - v0[0]*dv[2];
    nr[2] = v0[0]*dv[1] - v0[1]*dv[0];
    {
      float l = sqrtf(nr[0]*nr[0]+nr[1]*nr[1]+nr[2]*nr[2]);
      float inv = 1.0f / fmaxf(l, 1e-12f);
      for (int c = 0; c < 3; ++c) nr[c] *= inv;
    }
    if (i > 0 && i < NN - 1) {
      for (int c = 0; c < 3; ++c) tv[c] = coords[(size_t)(i+1)*3+c] - coords[(size_t)(i-1)*3+c];
    } else { for (int c = 0; c < 3; ++c) tv[c] = 0.0f; }
    {
      float l = sqrtf(tv[0]*tv[0]+tv[1]*tv[1]+tv[2]*tv[2]);
      float inv = 1.0f / fmaxf(l, 1e-12f);
      for (int c = 0; c < 3; ++c) tv[c] *= inv;
    }
    float* o = nv + (size_t)i * 12;
    for (int c = 0; c < 3; ++c) { o[c] = v0[c]; o[3+c] = dv[c]; o[6+c] = nr[c]; o[9+c] = tv[c]; }
  }
}

__global__ void k_build_edge(const float* __restrict__ ea, h16* __restrict__ ein) {
  int e = blockIdx.x * blockDim.x + threadIdx.x;
  if (e >= NE) return;
#pragma unroll
  for (int j = 0; j < 32; ++j) {
    float v = (j < 2) ? ea[(size_t)e * 35 + j] : ea[(size_t)e * 35 + 3 + j];
    ein[(size_t)e * 32 + j] = (h16)v;
  }
}

// ---------------- WMMA GEMM kernels ----------------------------------------
// node projection: [N,1312]f16 x W[128][1312]f16 -> node_s f32 + f16 copy
__global__ void k_nodeproj(const h16* __restrict__ nin, const h16* __restrict__ w,
                           const float* __restrict__ bias, float* __restrict__ ns,
                           h16* __restrict__ nsh) {
  const int tid = threadIdx.x, i0 = blockIdx.x * 16;
  const int wv = tid >> 5, lane = tid & 31, hi = lane >> 4, ln = lane & 15;
  const int n = wv * 16 + ln;
  const h16* arow = nin + (size_t)(i0 + ln) * KNODE;
  v8f acc = {};
  for (int kb = 0; kb < KNODE / 32; ++kb) {
    int koff = kb * 32;
    __builtin_prefetch(arow + koff + 64, 0, 3);
    acc = wmma32(ldA(arow, koff, hi), ldB(w, n, KNODE, koff, hi), acc);
  }
  float bb = bias[n];
#pragma unroll
  for (int r = 0; r < 8; ++r) {
    int node = i0 + r + hi * 8;
    float v = acc[r] + bb;
    ns[(size_t)node * H + n] = v;
    nsh[(size_t)node * H + n] = (h16)v;
  }
}

// edge projection: [E,32]f16 x W[128][32]f16 -> edge_s f16
__global__ void k_edgeproj(const h16* __restrict__ ein, const h16* __restrict__ w,
                           const float* __restrict__ bias, h16* __restrict__ esh) {
  const int tid = threadIdx.x, e0 = blockIdx.x * 16;
  const int wv = tid >> 5, lane = tid & 31, hi = lane >> 4, ln = lane & 15;
  const int n = wv * 16 + ln;
  v8f acc = {};
  acc = wmma32(ldA(ein + (size_t)(e0 + ln) * 32, 0, hi), ldB(w, n, 32, 0, hi), acc);
  float bb = bias[n];
#pragma unroll
  for (int r = 0; r < 8; ++r)
    esh[(size_t)(e0 + r + hi * 8) * H + n] = (h16)(acc[r] + bb);
}

// message GVP scalar path: gathered A = [ns[src] | ns[dst] | es[e]] (K=384),
// + vnorm@Wvs^T, LayerNorm, GELU, atomic scatter-add into aggr_s[dst]
__global__ void k_msg_gemm(const h16* __restrict__ nsh, const h16* __restrict__ esh,
                           const float* __restrict__ vn, const int* __restrict__ ei,
                           const h16* __restrict__ wss, const float* __restrict__ wvs,
                           const float* __restrict__ g, const float* __restrict__ b,
                           float* __restrict__ aggr_s) {
  __shared__ int srcl[16], dstl[16];
  __shared__ float vns[16][9];
  __shared__ float smem[16][H];
  __shared__ float mu[16], rsd[16];
  const int tid = threadIdx.x, e0 = blockIdx.x * 16;
  if (tid < 16) { srcl[tid] = ei[e0 + tid]; dstl[tid] = ei[NE + e0 + tid]; }
  if (tid < 144) vns[tid / 9][tid % 9] = vn[(size_t)(e0 + tid / 9) * 9 + tid % 9];
  __syncthreads();
  const int wv = tid >> 5, lane = tid & 31, hi = lane >> 4, ln = lane & 15;
  const int n = wv * 16 + ln;
  v8f acc = {};
#pragma unroll
  for (int kb = 0; kb < 12; ++kb) {
    const int seg = kb >> 2, koff = (kb & 3) * 32;
    const h16* arow;
    if (seg == 0)      arow = nsh + (size_t)srcl[ln] * H;
    else if (seg == 1) arow = nsh + (size_t)dstl[ln] * H;
    else               arow = esh + (size_t)(e0 + ln) * H;
    __builtin_prefetch(wss + (size_t)n * KMSG + koff + 128, 0, 3);
    acc = wmma32(ldA(arow, koff, hi), ldB(wss, n, KMSG, koff, hi), acc);
  }
  float wvr[9];
#pragma unroll
  for (int i = 0; i < 9; ++i) wvr[i] = wvs[n * 9 + i];
#pragma unroll
  for (int r = 0; r < 8; ++r) {
    int m = r + hi * 8;
    float add = 0.0f;
#pragma unroll
    for (int i = 0; i < 9; ++i) add += vns[m][i] * wvr[i];
    smem[m][n] = acc[r] + add;
  }
  __syncthreads();
  if (tid < 16) {
    float s = 0.0f, s2 = 0.0f;
    for (int c = 0; c < H; ++c) { float x = smem[tid][c]; s += x; s2 += x * x; }
    float m_ = s * (1.0f / H);
    mu[tid] = m_;
    rsd[tid] = rsqrtf(fmaxf(s2 * (1.0f / H) - m_ * m_, 0.0f) + 1e-5f);
  }
  __syncthreads();
#pragma unroll
  for (int j = 0; j < 8; ++j) {
    int idx = tid * 8 + j, row = idx >> 7, col = idx & 127;
    float x = (smem[row][col] - mu[row]) * rsd[row] * g[col] + b[col];
    x = gelu_exact(x);
    atomicAdd(&aggr_s[(size_t)dstl[row] * H + col], x);
  }
}

// update GVP scalar path: A = [ns | aggr_s] (K=256), + vnorm@Wvs^T, LN, GELU,
// residual add into node_s (f32) and node_s_h (f16)
__global__ void k_upd_gemm(h16* __restrict__ nsh, const h16* __restrict__ aggrh,
                           const float* __restrict__ vnu,
                           const h16* __restrict__ wss, const float* __restrict__ wvs,
                           const float* __restrict__ g, const float* __restrict__ b,
                           float* __restrict__ ns) {
  __shared__ float vns[16][8];
  __shared__ float smem[16][H];
  __shared__ float mu[16], rsd[16];
  const int tid = threadIdx.x, i0 = blockIdx.x * 16;
  if (tid < 128) vns[tid / 8][tid % 8] = vnu[(size_t)(i0 + tid / 8) * 8 + tid % 8];
  __syncthreads();
  const int wv = tid >> 5, lane = tid & 31, hi = lane >> 4, ln = lane & 15;
  const int n = wv * 16 + ln;
  v8f acc = {};
#pragma unroll
  for (int kb = 0; kb < 8; ++kb) {
    const int seg = kb >> 2, koff = (kb & 3) * 32;
    const h16* arow = (seg == 0 ? (const h16*)nsh : aggrh) + (size_t)(i0 + ln) * H;
    acc = wmma32(ldA(arow, koff, hi), ldB(wss, n, KUPD, koff, hi), acc);
  }
  float wvr[8];
#pragma unroll
  for (int i = 0; i < 8; ++i) wvr[i] = wvs[n * 8 + i];
#pragma unroll
  for (int r = 0; r < 8; ++r) {
    int m = r + hi * 8;
    float add = 0.0f;
#pragma unroll
    for (int i = 0; i < 8; ++i) add += vns[m][i] * wvr[i];
    smem[m][n] = acc[r] + add;
  }
  __syncthreads();
  if (tid < 16) {
    float s = 0.0f, s2 = 0.0f;
    for (int c = 0; c < H; ++c) { float x = smem[tid][c]; s += x; s2 += x * x; }
    float m_ = s * (1.0f / H);
    mu[tid] = m_;
    rsd[tid] = rsqrtf(fmaxf(s2 * (1.0f / H) - m_ * m_, 0.0f) + 1e-5f);
  }
  __syncthreads();
#pragma unroll
  for (int j = 0; j < 8; ++j) {
    int idx = tid * 8 + j, row = idx >> 7, col = idx & 127;
    float x = (smem[row][col] - mu[row]) * rsd[row] * g[col] + b[col];
    x = gelu_exact(x);
    size_t o = (size_t)(i0 + row) * H + col;
    float out = ns[o] + x;
    ns[o] = out;
    nsh[o] = (h16)out;
  }
}

// output head: updated = node_s @ node_out.W^T + b  ([N,128]x[128,1280])
__global__ void k_nodeout(const h16* __restrict__ nsh, const h16* __restrict__ w,
                          const float* __restrict__ bias, float* __restrict__ out) {
  const int tid = threadIdx.x, i0 = blockIdx.x * 16, nb = blockIdx.y * 128;
  const int wv = tid >> 5, lane = tid & 31, hi = lane >> 4, ln = lane & 15;
  const int n = nb + wv * 16 + ln;
  const h16* arow = nsh + (size_t)(i0 + ln) * H;
  v8f acc = {};
#pragma unroll
  for (int kb = 0; kb < 4; ++kb)
    acc = wmma32(ldA(arow, kb * 32, hi), ldB(w, n, H, kb * 32, hi), acc);
  float bb = bias[n];
#pragma unroll
  for (int r = 0; r < 8; ++r)
    out[(size_t)(i0 + r + hi * 8) * ESM + n] = acc[r] + bb;
}

// graph head: gf = node_s @ graph_out.W^T + b, atomic segment-sum by batch
__global__ void k_graphout(const h16* __restrict__ nsh, const h16* __restrict__ w,
                           const float* __restrict__ bias, const int* __restrict__ batch,
                           float* __restrict__ gsum) {
  __shared__ int bl[16];
  const int tid = threadIdx.x, i0 = blockIdx.x * 16, nb = blockIdx.y * 128;
  if (tid < 16) bl[tid] = batch[i0 + tid];
  __syncthreads();
  const int wv = tid >> 5, lane = tid & 31, hi = lane >> 4, ln = lane & 15;
  const int n = nb + wv * 16 + ln;
  const h16* arow = nsh + (size_t)(i0 + ln) * H;
  v8f acc = {};
#pragma unroll
  for (int kb = 0; kb < 4; ++kb)
    acc = wmma32(ldA(arow, kb * 32, hi), ldB(w, n, H, kb * 32, hi), acc);
  float bb = bias[n];
#pragma unroll
  for (int r = 0; r < 8; ++r)
    atomicAdd(&gsum[(size_t)bl[r + hi * 8] * OUTD + n], acc[r] + bb);
}

// ---------------- VALU side-channel kernels --------------------------------
// per-edge: vector norms, sigmoid gate (Wsv 4x384), Wvv 4x9, scatter aggr_v
__global__ void k_edge_vec(const h16* __restrict__ nsh, const h16* __restrict__ esh,
                           const float* __restrict__ nv, const float* __restrict__ ea,
                           const int* __restrict__ ei, const float* __restrict__ wsv,
                           const float* __restrict__ wvv, float* __restrict__ vn,
                           float* __restrict__ aggr_v) {
  int e = blockIdx.x * blockDim.x + threadIdx.x;
  if (e >= NE) return;
  int s = ei[e], d2 = ei[NE + e];
  float mv[9][3];
#pragma unroll
  for (int j = 0; j < 4; ++j)
#pragma unroll
    for (int c = 0; c < 3; ++c) {
      mv[j][c] = nv[(size_t)s * 12 + j * 3 + c];
      mv[4 + j][c] = nv[(size_t)d2 * 12 + j * 3 + c];
    }
#pragma unroll
  for (int c = 0; c < 3; ++c) mv[8][c] = ea[(size_t)e * 35 + 2 + c];
#pragma unroll
  for (int j = 0; j < 9; ++j)
    vn[(size_t)e * 9 + j] = sqrtf(mv[j][0]*mv[j][0] + mv[j][1]*mv[j][1] + mv[j][2]*mv[j][2]);
  float acc[4] = {0.f, 0.f, 0.f, 0.f};
  for (int k = 0; k < H; ++k) {
    float xs = (float)nsh[(size_t)s * H + k];
    float xd = (float)nsh[(size_t)d2 * H + k];
    float xe = (float)esh[(size_t)e * H + k];
#pragma unroll
    for (int o = 0; o < 4; ++o)
      acc[o] += xs * wsv[o * KMSG + k] + xd * wsv[o * KMSG + H + k] +
                xe * wsv[o * KMSG + 2 * H + k];
  }
#pragma unroll
  for (int o = 0; o < 4; ++o) {
    float gg = sigm(acc[o]);
#pragma unroll
    for (int c = 0; c < 3; ++c) {
      float vo = 0.0f;
#pragma unroll
      for (int i = 0; i < 9; ++i) vo += wvv[o * 9 + i] * mv[i][c];
      atomicAdd(&aggr_v[(size_t)d2 * 12 + o * 3 + c], vo * gg);
    }
  }
}

// per-node: update-GVP vector norms, gate (Wsv 4x256), Wvv 4x8,
// node_v residual update, aggr_s -> f16 copy for WMMA A operand
__global__ void k_node_mid(const h16* __restrict__ nsh, const float* __restrict__ aggr_s,
                           const float* __restrict__ aggr_v, float* __restrict__ nv,
                           float* __restrict__ vnu, h16* __restrict__ aggrh,
                           const float* __restrict__ wsv, const float* __restrict__ wvv) {
  int i = blockIdx.x * blockDim.x + threadIdx.x;
  if (i >= NN) return;
  float uv[8][3];
#pragma unroll
  for (int j = 0; j < 4; ++j)
#pragma unroll
    for (int c = 0; c < 3; ++c) {
      uv[j][c] = nv[(size_t)i * 12 + j * 3 + c];
      uv[4 + j][c] = aggr_v[(size_t)i * 12 + j * 3 + c];
    }
#pragma unroll
  for (int j = 0; j < 8; ++j)
    vnu[(size_t)i * 8 + j] = sqrtf(uv[j][0]*uv[j][0] + uv[j][1]*uv[j][1] + uv[j][2]*uv[j][2]);
  float acc[4] = {0.f, 0.f, 0.f, 0.f};
  for (int k = 0; k < H; ++k) {
    float xs = (float)nsh[(size_t)i * H + k];
    float xa = aggr_s[(size_t)i * H + k];
    aggrh[(size_t)i * H + k] = (h16)xa;
#pragma unroll
    for (int o = 0; o < 4; ++o)
      acc[o] += xs * wsv[o * KUPD + k] + xa * wsv[o * KUPD + H + k];
  }
#pragma unroll
  for (int o = 0; o < 4; ++o) {
    float gg = sigm(acc[o]);
#pragma unroll
    for (int c = 0; c < 3; ++c) {
      float vo = 0.0f;
#pragma unroll
      for (int j = 0; j < 8; ++j) vo += wvv[o * 8 + j] * uv[j][c];
      nv[(size_t)i * 12 + o * 3 + c] += vo * gg;
    }
  }
}

__global__ void k_counts(const int* __restrict__ batch, float* __restrict__ cnt) {
  int i = blockIdx.x * blockDim.x + threadIdx.x;
  if (i < NN) atomicAdd(&cnt[batch[i]], 1.0f);
}

__global__ void k_final(const float* __restrict__ gsum, const float* __restrict__ cnt,
                        float* __restrict__ out) {
  int idx = blockIdx.x * blockDim.x + threadIdx.x;  // 32*256
  int g = idx >> 8;
  out[idx] = gsum[idx] / fmaxf(cnt[g], 1.0f);
}

// ---------------------------------------------------------------------------
extern "C" void kernel_launch(void* const* d_in, const int* in_sizes, int n_in,
                              void* d_out, int out_size, void* d_ws, size_t ws_size,
                              hipStream_t stream) {
  (void)out_size; (void)ws_size;
  auto find1 = [&](int v, int from) -> int {
    for (int i = from; i < n_in; ++i) if (in_sizes[i] == v) return i;
    return 0;
  };
  const int i_esm = find1(NN * ESM, 0);
  const int i_geo = find1(NN * GEO, 0);
  const int i_co  = find1(NN * 3, 0);
  const int i_ea  = find1(NE * 35, 0);
  const int i_ei  = find1(2 * NE, 0);
  const int i_bi  = find1(NN, 0);
  const int i_npW = find1(H * 1289, 0);
  const int i_epW = find1(H * 32, 0);
  const int i_noW = find1(ESM * H, 0);
  int i_goW = 0;
  for (int i = 0; i + 1 < n_in; ++i)
    if (in_sizes[i] == OUTD * H && in_sizes[i + 1] == OUTD) { i_goW = i; break; }
  int lbase[3]; int p = 0;
  for (int l = 0; l < 3; ++l) { lbase[l] = find1(H * KMSG, p); p = lbase[l] + 1; }
  // detect flatten order of GVP dicts: sorted keys -> Wss,Wsv,Wvs,Wvv,b,g
  const bool sorted = (in_sizes[lbase[0] + 1] == 4 * KMSG);

  auto F = [&](int i) { return (const float*)d_in[i]; };
  const float* esm = F(i_esm); const float* geo = F(i_geo); const float* co = F(i_co);
  const float* ea = F(i_ea);
  const int* ei = (const int*)d_in[i_ei];
  const int* bi = (const int*)d_in[i_bi];

  struct GvpP { const float *Wss, *Wsv, *Wvs, *Wvv, *g, *b; };
  GvpP msg[3], upd[3];
  for (int l = 0; l < 3; ++l) {
    int base = lbase[l];
    if (sorted) {
      msg[l] = { F(base), F(base + 1), F(base + 2), F(base + 3), F(base + 5), F(base + 4) };
      upd[l] = { F(base + 6), F(base + 7), F(base + 8), F(base + 9), F(base + 11), F(base + 10) };
    } else {  // insertion order: Wss,Wvs,Wsv,Wvv,g,b
      msg[l] = { F(base), F(base + 2), F(base + 1), F(base + 3), F(base + 4), F(base + 5) };
      upd[l] = { F(base + 6), F(base + 8), F(base + 7), F(base + 9), F(base + 10), F(base + 11) };
    }
  }

  // ---- workspace carve-up ----
  char* wsp = (char*)d_ws;
  size_t off = 0;
  auto carve = [&](size_t bytes) -> void* {
    void* r = wsp + off;
    off = (off + bytes + 255) & ~(size_t)255;
    return r;
  };
  float* ns     = (float*)carve((size_t)NN * H * 4);
  h16*   nsh    = (h16*)  carve((size_t)NN * H * 2);
  float* nv     = (float*)carve((size_t)NN * 12 * 4);
  h16*   nin    = (h16*)  carve((size_t)NN * KNODE * 2);
  h16*   ein    = (h16*)  carve((size_t)NE * 32 * 2);
  h16*   esh    = (h16*)  carve((size_t)NE * H * 2);
  float* vn     = (float*)carve((size_t)NE * 9 * 4);
  float* vnu    = (float*)carve((size_t)NN * 8 * 4);
  float* aggr_s = (float*)carve((size_t)NN * H * 4);
  float* aggr_v = (float*)carve((size_t)NN * 12 * 4);
  h16*   aggrh  = (h16*)  carve((size_t)NN * H * 2);
  float* gsum   = (float*)carve((size_t)NGR * OUTD * 4);
  float* cnt    = (float*)carve((size_t)NGR * 4);
  h16* wnp = (h16*)carve((size_t)H * KNODE * 2);
  h16* wep = (h16*)carve((size_t)H * 32 * 2);
  h16* wms[3], *wup[3];
  for (int l = 0; l < 3; ++l) wms[l] = (h16*)carve((size_t)H * KMSG * 2);
  for (int l = 0; l < 3; ++l) wup[l] = (h16*)carve((size_t)H * KUPD * 2);
  h16* wno = (h16*)carve((size_t)ESM * H * 2);
  h16* wgo = (h16*)carve((size_t)OUTD * H * 2);

  const int B = 256;
  auto pack = [&](const float* s, h16* d, int O, int Ks, int Kd) {
    int tot = O * Kd;
    k_pack<<<(tot + B - 1) / B, B, 0, stream>>>(s, d, O, Ks, Kd);
  };
  pack(F(i_npW), wnp, H, 1289, KNODE);
  pack(F(i_epW), wep, H, 32, 32);
  for (int l = 0; l < 3; ++l) { pack(msg[l].Wss, wms[l], H, KMSG, KMSG); pack(upd[l].Wss, wup[l], H, KUPD, KUPD); }
  pack(F(i_noW), wno, ESM, H, H);
  pack(F(i_goW), wgo, OUTD, H, H);

  k_build_node<<<NN, 128, 0, stream>>>(esm, geo, co, nin, nv);
  k_build_edge<<<NE / B, B, 0, stream>>>(ea, ein);
  k_nodeproj<<<NN / 16, B, 0, stream>>>(nin, wnp, F(i_npW + 1), ns, nsh);
  k_edgeproj<<<NE / 16, B, 0, stream>>>(ein, wep, F(i_epW + 1), esh);

  for (int l = 0; l < 3; ++l) {
    k_zero<<<(NN * H + B - 1) / B, B, 0, stream>>>(aggr_s, NN * H);
    k_zero<<<(NN * 12 + B - 1) / B, B, 0, stream>>>(aggr_v, NN * 12);
    k_edge_vec<<<NE / B, B, 0, stream>>>(nsh, esh, nv, ea, ei, msg[l].Wsv, msg[l].Wvv, vn, aggr_v);
    k_msg_gemm<<<NE / 16, B, 0, stream>>>(nsh, esh, vn, ei, wms[l], msg[l].Wvs,
                                          msg[l].g, msg[l].b, aggr_s);
    k_node_mid<<<(NN + B - 1) / B, B, 0, stream>>>(nsh, aggr_s, aggr_v, nv, vnu, aggrh,
                                                   upd[l].Wsv, upd[l].Wvv);
    k_upd_gemm<<<NN / 16, B, 0, stream>>>(nsh, aggrh, vnu, wup[l], upd[l].Wvs,
                                          upd[l].g, upd[l].b, ns);
  }

  float* out = (float*)d_out;
  k_zero<<<(NGR * OUTD + B - 1) / B, B, 0, stream>>>(gsum, NGR * OUTD);
  k_zero<<<1, 32, 0, stream>>>(cnt, NGR);
  k_counts<<<(NN + B - 1) / B, B, 0, stream>>>(bi, cnt);
  k_graphout<<<dim3(NN / 16, 2), B, 0, stream>>>(nsh, wgo, F(i_goW + 1), bi, gsum);
  k_final<<<NGR, B, 0, stream>>>(gsum, cnt, out);
  k_nodeout<<<dim3(NN / 16, ESM / 128), B, 0, stream>>>(nsh, wno, F(i_noW + 1), out + NGR * OUTD);
}
